// Theom_logit_77094662963449
// MI455X (gfx1250) — matrix operand order repaired and checked by reading, not compiled, not run on hardware
//
#include <hip/hip_runtime.h>
#include <hip/hip_bf16.h>

typedef __attribute__((ext_vector_type(16))) __bf16 v16bf;
typedef __attribute__((ext_vector_type(8)))  float  v8f;
typedef __attribute__((ext_vector_type(4)))  unsigned int v4u;

#define G_  128
#define T_  512
#define E_  1024
#define H1_ 1024
#define H2_ 512

union ABFrag {
    v16bf v;
    v4u   q[2];
};

__device__ __forceinline__ unsigned short f2bf(float f) {
    unsigned int u = __float_as_uint(f);
    u += 0x7FFFu + ((u >> 16) & 1u);   // round-to-nearest-even
    return (unsigned short)(u >> 16);
}

__device__ __forceinline__ float wave_sum(float x) {
    x += __shfl_xor(x, 16, 32);
    x += __shfl_xor(x, 8, 32);
    x += __shfl_xor(x, 4, 32);
    x += __shfl_xor(x, 2, 32);
    x += __shfl_xor(x, 1, 32);
    return x;
}

// ---------------- prep 1: hg = goal @ Wg.T , ht = thm @ Wt.T ----------------
__global__ void prep_hght(const float* __restrict__ goal, const float* __restrict__ thm,
                          const float* __restrict__ W1,
                          float* __restrict__ hg, float* __restrict__ ht) {
    int idx = blockIdx.x * 256 + threadIdx.x;    // 640*1024 threads
    int row = idx >> 10;
    int h   = idx & 1023;
    const float* x;
    const float* w;
    if (row < G_) { x = goal + row * E_;        w = W1 + (size_t)h * 3072; }
    else          { x = thm + (row - G_) * E_;  w = W1 + (size_t)h * 3072 + E_; }
    float acc = 0.f;
    for (int e = 0; e < E_; e += 4) {
        float4 xv = *(const float4*)(x + e);
        float4 wv = *(const float4*)(w + e);
        acc += xv.x * wv.x + xv.y * wv.y + xv.z * wv.z + xv.w * wv.w;
    }
    if (row < G_) hg[idx] = acc;
    else          ht[idx - G_ * E_] = acc;
}

// ---------------- prep 2: bf16 conversion of Wm and W2 ----------------
__global__ void prep_cvt(const float* __restrict__ W1, const float* __restrict__ W2,
                         unsigned short* __restrict__ Wm_bf, unsigned short* __restrict__ W2_bf) {
    int idx = blockIdx.x * 256 + threadIdx.x;    // (1024+512)*1024 threads
    if (idx < H1_ * E_) {
        int h = idx >> 10, e = idx & 1023;
        Wm_bf[idx] = f2bf(W1[(size_t)h * 3072 + 2048 + e]);
    } else {
        int j = idx - H1_ * E_;
        W2_bf[j] = f2bf(W2[j]);
    }
}

// ---------------- fused main kernel ----------------
// grid: (T/32, G), block 512 (16 waves). dynamic LDS = 192KB:
//   [0,64K)    : p_lds  bf16 32x1024  (reused as a2_lds after GEMM1)
//   [64K,192K) : h1_lds f32  32x1024  (first 64K reused as h2_lds f32 32x512)
// Each wave register-blocks 2 M-tiles x 4 N-tiles (8 accumulators) in GEMM1
// and 2 M-tiles x 2 N-tiles (4 accumulators) in GEMM2. All A/B fragments of a
// K-step are loaded into distinct live registers BEFORE the WMMA burst so the
// loads clause together and overlap the previous iteration's WMMAs.
__global__ void __launch_bounds__(512) fused(
    const float* __restrict__ goal, const float* __restrict__ thm,
    const float* __restrict__ b1,  const float* __restrict__ ln1g, const float* __restrict__ ln1b,
    const float* __restrict__ b2,  const float* __restrict__ ln2g, const float* __restrict__ ln2b,
    const float* __restrict__ W3,  const float* __restrict__ b3,
    const float* __restrict__ hg,  const float* __restrict__ ht,
    const unsigned short* __restrict__ Wm_bf, const unsigned short* __restrict__ W2_bf,
    float* __restrict__ out)
{
    extern __shared__ char smem[];
    unsigned short* p_lds  = (unsigned short*)smem;          // 32x1024 bf16
    unsigned short* a2_lds = p_lds;                          // alias (p dead after GEMM1)
    float* h1_lds = (float*)(smem + 64 * 1024);              // 32x1024 f32
    float* h2_lds = (float*)(smem + 64 * 1024);              // alias (h1 dead after LN1)

    const int tid  = threadIdx.x;
    const int g    = blockIdx.y;
    const int t0   = blockIdx.x * 32;
    const int wv   = tid >> 5;
    const int lane = tid & 31;

    // ---- Phase 0: p[m,k] = goal[g,k] * thm[t0+m,k]  -> bf16 LDS ----
    {
        const float* grow = goal + (size_t)g * E_;
        for (int i = 0; i < 64; ++i) {
            int f = i * 512 + tid;              // 32*1024 elements
            int m = f >> 10, k = f & 1023;
            float pv = grow[k] * thm[(size_t)(t0 + m) * E_ + k];
            p_lds[f] = f2bf(pv);
        }
    }
    __syncthreads();

    const int nlane = lane & 15;                 // A row / B col / D col for this lane
    const int offA  = (lane < 16) ? 0 : 8;       // 16-bit A layout K-offset
    const int baseB = (lane < 16) ? 0 : 16;      // unused; B loads use offA, same k order as A
    const int mbase = (lane < 16) ? 0 : 8;       // C/D row base

    // ---- Phase 1: GEMM1  h1[m,h] = sum_k p[m,k]*Wm[h,k] ----
    {
        const int h0 = wv * 64;                  // this wave owns 4 N-tiles
        v8f acc[8];
#pragma unroll
        for (int i = 0; i < 8; ++i) acc[i] = (v8f){0.f,0.f,0.f,0.f,0.f,0.f,0.f,0.f};
        const unsigned short* arow0 = p_lds + nlane * E_;         // M-tile 0
        const unsigned short* arow1 = p_lds + (16 + nlane) * E_;  // M-tile 1
        const unsigned short* brow0 = Wm_bf + (size_t)(h0 + nlane) * E_;
        __builtin_prefetch(brow0, 0, 1);
        for (int kb = 0; kb < E_; kb += 32) {
            // gather ALL fragments for this K-step into distinct live registers
            ABFrag a0, a1, b[4];
            a0.q[0] = *(const v4u*)(arow0 + kb + offA);
            a0.q[1] = *(const v4u*)(arow0 + kb + 16 + offA);
            a1.q[0] = *(const v4u*)(arow1 + kb + offA);
            a1.q[1] = *(const v4u*)(arow1 + kb + 16 + offA);
#pragma unroll
            for (int nt = 0; nt < 4; ++nt) {
                const unsigned short* brow = brow0 + (size_t)(nt * 16) * E_;
                b[nt].q[0] = *(const v4u*)(brow + kb + offA);
                b[nt].q[1] = *(const v4u*)(brow + kb + 16 + offA);
            }
            // WMMA burst: 8 independent accumulator chains
#pragma unroll
            for (int nt = 0; nt < 4; ++nt) {
                acc[nt] = __builtin_amdgcn_wmma_f32_16x16x32_bf16(
                    false, a0.v, false, b[nt].v, (short)0, acc[nt], false, false);
                acc[4 + nt] = __builtin_amdgcn_wmma_f32_16x16x32_bf16(
                    false, a1.v, false, b[nt].v, (short)0, acc[4 + nt], false, false);
            }
        }
#pragma unroll
        for (int nt = 0; nt < 4; ++nt) {
#pragma unroll
            for (int r = 0; r < 8; ++r) {
                h1_lds[(mbase + r) * H1_ + h0 + nt * 16 + nlane]      = acc[nt][r];
                h1_lds[(16 + mbase + r) * H1_ + h0 + nt * 16 + nlane] = acc[4 + nt][r];
            }
        }
    }
    __syncthreads();

    // ---- Phase 2: x = h1 + hg + ht + b1 ; LN1 ; ReLU ; -> bf16 a2 ----
    for (int mm = 0; mm < 2; ++mm) {
        int m = wv * 2 + mm;                     // 16 waves x 2 = 32 rows
        int t = t0 + m;
        float s = 0.f, s2 = 0.f;
        for (int j = 0; j < 32; ++j) {
            int h = lane + j * 32;
            float x = h1_lds[m * H1_ + h] + hg[(size_t)g * E_ + h]
                    + ht[(size_t)t * E_ + h] + b1[h];
            h1_lds[m * H1_ + h] = x;
            s += x; s2 += x * x;
        }
        s  = wave_sum(s);
        s2 = wave_sum(s2);
        float mu   = s * (1.f / 1024.f);
        float var  = s2 * (1.f / 1024.f) - mu * mu;
        float rstd = rsqrtf(var + 1e-5f);
        for (int j = 0; j < 32; ++j) {
            int h = lane + j * 32;
            float x = h1_lds[m * H1_ + h];
            float y = (x - mu) * rstd * ln1g[h] + ln1b[h];
            y = fmaxf(y, 0.f);
            a2_lds[m * H1_ + h] = f2bf(y);
        }
    }
    __syncthreads();

    // ---- Phase 3: GEMM2  h2[m,j] = sum_k a2[m,k]*W2[j,k] ----
    {
        const int j0 = wv * 32;                  // this wave owns 2 N-tiles
        v8f acc[4];
#pragma unroll
        for (int i = 0; i < 4; ++i) acc[i] = (v8f){0.f,0.f,0.f,0.f,0.f,0.f,0.f,0.f};
        const unsigned short* arow0 = a2_lds + nlane * H1_;
        const unsigned short* arow1 = a2_lds + (16 + nlane) * H1_;
        const unsigned short* brow0 = W2_bf + (size_t)(j0 + nlane) * H1_;
        __builtin_prefetch(brow0, 0, 1);
        for (int kb = 0; kb < H1_; kb += 32) {
            ABFrag a0, a1, b[2];
            a0.q[0] = *(const v4u*)(arow0 + kb + offA);
            a0.q[1] = *(const v4u*)(arow0 + kb + 16 + offA);
            a1.q[0] = *(const v4u*)(arow1 + kb + offA);
            a1.q[1] = *(const v4u*)(arow1 + kb + 16 + offA);
#pragma unroll
            for (int nt = 0; nt < 2; ++nt) {
                const unsigned short* brow = brow0 + (size_t)(nt * 16) * H1_;
                b[nt].q[0] = *(const v4u*)(brow + kb + offA);
                b[nt].q[1] = *(const v4u*)(brow + kb + 16 + offA);
            }
#pragma unroll
            for (int nt = 0; nt < 2; ++nt) {
                acc[nt] = __builtin_amdgcn_wmma_f32_16x16x32_bf16(
                    false, a0.v, false, b[nt].v, (short)0, acc[nt], false, false);
                acc[2 + nt] = __builtin_amdgcn_wmma_f32_16x16x32_bf16(
                    false, a1.v, false, b[nt].v, (short)0, acc[2 + nt], false, false);
            }
        }
#pragma unroll
        for (int nt = 0; nt < 2; ++nt) {
#pragma unroll
            for (int r = 0; r < 8; ++r) {
                h2_lds[(mbase + r) * H2_ + j0 + nt * 16 + nlane]      = acc[nt][r];
                h2_lds[(16 + mbase + r) * H2_ + j0 + nt * 16 + nlane] = acc[2 + nt][r];
            }
        }
    }
    __syncthreads();

    // ---- Phase 4: +b2 ; LN2 ; ReLU ; dot W3 ; +b3 -> logits ----
    for (int mm = 0; mm < 2; ++mm) {
        int m = wv * 2 + mm;                     // 32 rows
        float xr[16];
        float s = 0.f, s2 = 0.f;
        for (int j = 0; j < 16; ++j) {
            int c = lane + j * 32;
            float x = h2_lds[m * H2_ + c] + b2[c];
            xr[j] = x; s += x; s2 += x * x;
        }
        s  = wave_sum(s);
        s2 = wave_sum(s2);
        float mu   = s * (1.f / 512.f);
        float var  = s2 * (1.f / 512.f) - mu * mu;
        float rstd = rsqrtf(var + 1e-5f);
        float dot = 0.f;
        for (int j = 0; j < 16; ++j) {
            int c = lane + j * 32;
            float y = (xr[j] - mu) * rstd * ln2g[c] + ln2b[c];
            y = fmaxf(y, 0.f);
            dot += y * W3[c];
        }
        dot = wave_sum(dot);
        if (lane == 0) out[(size_t)g * T_ + t0 + m] = dot + b3[0];
    }
}

extern "C" void kernel_launch(void* const* d_in, const int* in_sizes, int n_in,
                              void* d_out, int out_size, void* d_ws, size_t ws_size,
                              hipStream_t stream) {
    const float* goal = (const float*)d_in[0];
    const float* thm  = (const float*)d_in[1];
    const float* W1   = (const float*)d_in[2];
    const float* b1   = (const float*)d_in[3];
    const float* ln1g = (const float*)d_in[4];
    const float* ln1b = (const float*)d_in[5];
    const float* W2   = (const float*)d_in[6];
    const float* b2   = (const float*)d_in[7];
    const float* ln2g = (const float*)d_in[8];
    const float* ln2b = (const float*)d_in[9];
    const float* W3   = (const float*)d_in[10];
    const float* b3   = (const float*)d_in[11];
    float* out = (float*)d_out;

    char* ws = (char*)d_ws;
    float*          hg    = (float*)(ws);                                      // 512 KB
    float*          ht    = (float*)(ws + (512u << 10));                       // 2 MB
    unsigned short* Wm_bf = (unsigned short*)(ws + (512u << 10) + (2u << 20)); // 2 MB
    unsigned short* W2_bf = (unsigned short*)(ws + (512u << 10) + (4u << 20)); // 1 MB

    prep_hght<<<(640 * 1024) / 256, 256, 0, stream>>>(goal, thm, W1, hg, ht);
    prep_cvt<<<(1536 * 1024) / 256, 256, 0, stream>>>(W1, W2, Wm_bf, W2_bf);

    dim3 grid(T_ / 32, G_);
    fused<<<grid, 512, 192 * 1024, stream>>>(goal, thm, b1, ln1g, ln1b,
                                             b2, ln2g, ln2b, W3, b3,
                                             hg, ht, Wm_bf, W2_bf, out);
}
